// Matcher_61332132986972
// MI455X (gfx1250) — matrix operand order repaired and checked by reading, not compiled, not run on hardware
//
#include <hip/hip_runtime.h>
#include <hip/hip_bf16.h>

// Problem dims (fixed by setup_inputs)
#define BATCH 64
#define NQ    256   // proposals (columns of LAP)
#define NG    64    // max GT (rows of LAP)
#define NC    768   // classes

typedef __attribute__((ext_vector_type(16))) _Float16 v16h;
typedef __attribute__((ext_vector_type(8)))  float    v8f;

// ---------------------------------------------------------------------------
// Kernel 1: WMMA einsum  class_mat = -(scores @ positive_map^T)
// Dead code in the reference (deleted before use) — implemented here to
// exercise the gfx1250 matrix path (v_wmma_f32_16x16x32_f16). Each wave owns
// one 16x16 output tile, marches K=768 in 32-wide steps, then writes a tile
// checksum to scratch (guarded by ws size) so the WMMAs are not DCE'd.
// Never touches d_out.
// ---------------------------------------------------------------------------
__global__ __launch_bounds__(32) void classmat_wmma_kernel(
    const float* __restrict__ scores,   // [B,Q,C]
    const float* __restrict__ pmap,     // [B,G,C]
    float* __restrict__ ws, unsigned ws_floats)
{
  const unsigned blk  = blockIdx.x;          // B * (Q/16) * (G/16) = 4096 blocks
  const unsigned gt   = blk & 3u;            // G tile (4)
  const unsigned qt   = (blk >> 2) & 15u;    // Q tile (16)
  const unsigned b    = blk >> 6;            // batch
  const unsigned lane = threadIdx.x;
  const unsigned hi   = lane >> 4;           // lane half (ISA A/B layouts)
  const unsigned mrow = lane & 15u;          // M (for A) / N (for B)

  const float* Arow = scores + ((size_t)b * NQ + qt * 16u + mrow) * NC;
  const float* Brow = pmap   + ((size_t)b * NG + gt * 16u + mrow) * NC;

  v8f c = {};
  for (int k0 = 0; k0 < NC; k0 += 32) {
    v16h a, bb;
    // A 16x32 f16 layout: lanes 0-15 M=lane; VGPR v<4 -> K=2v+8*hi,
    // v>=4 -> K=16+2(v-4)+8*hi  (CDNA5 ISA §7.12.2)
#pragma unroll
    for (int v = 0; v < 8; ++v) {
      int kb = k0 + ((v < 4) ? (2 * v + 8 * (int)hi)
                             : (16 + 2 * (v - 4) + 8 * (int)hi));
      a[2 * v]     = (_Float16)Arow[kb];
      a[2 * v + 1] = (_Float16)Arow[kb + 1];
    }
    // B 32x16 f16 layout: N = lane%16, same K order as A (r<4 -> K=2r+8*hi, r>=4 -> K=16+2(r-4)+8*hi)
#pragma unroll
    for (int r = 0; r < 8; ++r) {
      int kb = k0 + ((r < 4) ? (2 * r + 8 * (int)hi)
                             : (16 + 2 * (r - 4) + 8 * (int)hi));
      bb[2 * r]     = (_Float16)Brow[kb];
      bb[2 * r + 1] = (_Float16)Brow[kb + 1];
    }
    // (neg_a, A, neg_b, B, c_mod, C, reuse_a, reuse_b)
    c = __builtin_amdgcn_wmma_f32_16x16x32_f16(false, a, false, bb,
                                               (short)0, c, false, false);
  }

  float s = 0.f;
#pragma unroll
  for (int r = 0; r < 8; ++r) s += c[r];
  for (int off = 16; off > 0; off >>= 1) s += __shfl_down(s, off, 32);
  if (lane == 0 && blk < ws_floats) ws[blk] = -s;   // class_mat is negated
}

// ---------------------------------------------------------------------------
// Kernel 2: per-batch Jonker-Volgenant LAP (exact mirror of _lsa_minrows).
// One workgroup (256 threads) per batch. Cost tile lives in LDS, loaded with
// gfx1250 async global->LDS b32 ops doing a transposing scatter into a
// padded [G][Q+1] layout (bank-conflict-free row reads). Dijkstra control is
// serial on lane 0; column relaxation and argmin are parallel over 256 lanes.
// Dual variables kept in f64 to match the reference's float64 tie-breaking.
// ---------------------------------------------------------------------------
__global__ __launch_bounds__(NQ) void hungarian_kernel(
    const float* __restrict__ gious,   // [B,Q,G]
    const int*   __restrict__ nact,    // [B]
    float* __restrict__ out_inds,      // [B,Q] (float-encoded int)
    float* __restrict__ out_mask)      // [B,Q]
{
  __shared__ float  cost[NG][NQ + 1];  // cost[g][q] holds raw gious[q][g]
  __shared__ double shortest[NQ];
  __shared__ double vdual[NQ];
  __shared__ double udual[NG];
  __shared__ int    path[NQ];
  __shared__ int    row4col[NQ];
  __shared__ int    col4row[NG];
  __shared__ int    SC[NQ];
  __shared__ int    SR[NG];
  __shared__ double red_v[NQ];
  __shared__ int    red_i[NQ];
  __shared__ int    ans_g[NQ];
  __shared__ float  ans_m[NQ];
  __shared__ double s_min;
  __shared__ int    s_i, s_sink;

  const int tid = threadIdx.x;
  const int b   = blockIdx.x;
  int nb = nact[b];
  if (nb < 0) nb = 0;
  if (nb > NG) nb = NG;

  // ---- async global->LDS load with transpose: LDS[g][q] = gious[b][q][g] ----
  {
    const float* gb = gious + (size_t)b * NQ * NG;
    // low 32 bits of a flat LDS pointer == wave-relative LDS byte offset
    unsigned lds_base = (unsigned)(size_t)(&cost[0][0]);
    for (int e = tid; e < NQ * NG; e += NQ) {
      unsigned q = (unsigned)e >> 6;     // e / 64
      unsigned g = (unsigned)e & 63u;    // e % 64
      unsigned lds_addr = lds_base + (g * (unsigned)(NQ + 1) + q) * 4u;
      unsigned voff = (unsigned)e * 4u;  // coalesced global source
      asm volatile("global_load_async_to_lds_b32 %0, %1, %2"
                   :: "v"(lds_addr), "v"(voff), "s"(gb) : "memory");
    }
    asm volatile("s_wait_asynccnt 0" ::: "memory");
  }

  // ---- global (per-batch) init ----
  row4col[tid] = -1;
  path[tid]    = -1;
  vdual[tid]   = 0.0;
  if (tid < NG) { udual[tid] = 0.0; col4row[tid] = -1; }
  __syncthreads();

  for (int cur = 0; cur < nb; ++cur) {
    shortest[tid] = 1e300;
    SC[tid] = 0;
    if (tid < NG) SR[tid] = 0;
    if (tid == 0) { s_i = cur; s_min = 0.0; s_sink = -1; }
    __syncthreads();

    while (true) {
      const int    i  = s_i;
      const double mv = s_min;
      if (tid == 0) SR[i] = 1;
      // relax all columns not yet in SC (cost = -2 * gious, exact in f64)
      if (!SC[tid]) {
        double r = ((mv + (-2.0 * (double)cost[i][tid])) - udual[i]) - vdual[tid];
        if (r < shortest[tid]) { shortest[tid] = r; path[tid] = i; }
      }
      red_v[tid] = SC[tid] ? 1e301 : shortest[tid];
      red_i[tid] = tid;
      __syncthreads();
      // argmin over remaining columns, tie-break = smallest column index
      for (int s = NQ / 2; s > 0; s >>= 1) {
        if (tid < s) {
          double v2 = red_v[tid + s];
          int    i2 = red_i[tid + s];
          if (v2 < red_v[tid] || (v2 == red_v[tid] && i2 < red_i[tid])) {
            red_v[tid] = v2; red_i[tid] = i2;
          }
        }
        __syncthreads();
      }
      if (tid == 0) {
        int j = red_i[0];
        s_min = red_v[0];
        SC[j] = 1;
        int rj = row4col[j];
        if (rj < 0) s_sink = j; else s_i = rj;
      }
      __syncthreads();
      if (s_sink >= 0) break;
    }

    // ---- dual variable update (parallel over rows/cols) ----
    {
      const double mv = s_min;
      if (tid == 0) udual[cur] += mv;
      if (tid < NG && tid != cur && SR[tid])
        udual[tid] += mv - shortest[col4row[tid]];
      if (SC[tid])
        vdual[tid] -= mv - shortest[tid];
    }
    __syncthreads();

    // ---- augment alternating path (serial, <= NG steps) ----
    if (tid == 0) {
      int j = s_sink;
      while (true) {
        int i = path[j];
        row4col[j] = i;
        int jn = col4row[i];
        col4row[i] = j;
        j = jn;
        if (i == cur) break;
      }
    }
    __syncthreads();
  }

  // ---- scatter results: per_prop_gt_inds / proposal_matched_mask ----
  ans_g[tid] = 0;
  ans_m[tid] = 0.f;
  __syncthreads();
  if (tid < nb) {
    int q = col4row[tid];
    ans_g[q] = tid;
    ans_m[q] = 1.f;
  }
  __syncthreads();
  out_inds[(size_t)b * NQ + tid] = (float)ans_g[tid];
  out_mask[(size_t)b * NQ + tid] = ans_m[tid];
}

// ---------------------------------------------------------------------------
extern "C" void kernel_launch(void* const* d_in, const int* in_sizes, int n_in,
                              void* d_out, int out_size, void* d_ws, size_t ws_size,
                              hipStream_t stream) {
  (void)in_sizes; (void)n_in; (void)out_size;
  const float* scores = (const float*)d_in[0];  // [B,Q,C] f32
  const float* gious  = (const float*)d_in[1];  // [B,Q,G] f32
  const float* pmap   = (const float*)d_in[2];  // [B,G,C] f32
  /* d_in[3] = sem_cls_label: unused by the reference's live path */
  const int*   nact   = (const int*)d_in[4];    // [B]

  float* out_inds = (float*)d_out;              // [B,Q]
  float* out_mask = out_inds + BATCH * NQ;      // [B,Q]

  // Live output path: 64 LAPs, one workgroup per batch.
  hungarian_kernel<<<BATCH, NQ, 0, stream>>>(gious, nact, out_inds, out_mask);

  // Dead-code einsum on the matrix engine (checksums to scratch only).
  unsigned ws_floats = (unsigned)(ws_size / sizeof(float));
  classmat_wmma_kernel<<<BATCH * (NQ / 16) * (NG / 16), 32, 0, stream>>>(
      scores, pmap, (float*)d_ws, ws_floats);
}
